// ConditionModulatedMambaBlock_72576357368411
// MI455X (gfx1250) — matrix operand order repaired and checked by reading, not compiled, not run on hardware
//
#include <hip/hip_runtime.h>
#include <hip/hip_bf16.h>

// ---------------------------------------------------------------------------
// Types for CDNA5 WMMA
// ---------------------------------------------------------------------------
typedef __attribute__((ext_vector_type(16))) _Float16 v16h;
typedef __attribute__((ext_vector_type(8)))  float    v8f;
typedef __attribute__((ext_vector_type(2)))  _Float16 h2v;

__device__ __forceinline__ float sigmoidf_(float x) { return 1.f / (1.f + __expf(-x)); }
__device__ __forceinline__ float siluf_(float x)    { return x * sigmoidf_(x); }

// ---------------------------------------------------------------------------
// Generic WMMA GEMM:  C[M,N] = A[M,K] @ W[N,K]^T (+ bias[N])
// A row-major f32, W row-major f32 (PyTorch-style weight), C row-major f32.
// Requirements: M % 128 == 0, K % 32 == 0.  N arbitrary (masked).
// Block: 256 threads = 8 waves.  Block tile: 128(M) x 64(N).
//   A tile (128x32 f32)  -> LDS via GLOBAL_LOAD_ASYNC_TO_LDS_B128 (ASYNCcnt)
//   W tile (64x32)       -> converted to f16 once at staging (shared by 8 waves)
// Each wave computes a 16(M) x 64(N) strip with 4 f32 accumulators.
// ---------------------------------------------------------------------------
__global__ __launch_bounds__(256) void gemm_wmma_kernel(
    const float* __restrict__ A, const float* __restrict__ W,
    const float* __restrict__ bias, float* __restrict__ C,
    int M, int K, int N)
{
  // stride 36 floats = 144 B: keeps 16B alignment for async b128 chunks and
  // breaks bank periodicity for the fragment gathers.
  __shared__ float    lAf[128][36];
  __shared__ _Float16 lBh[64][36];

  const int m0   = blockIdx.x * 128;
  const int n0   = blockIdx.y * 64;
  const int tid  = threadIdx.x;
  const int wave = tid >> 5;
  const int lane = tid & 31;
  const int hlf  = lane >> 4;
  const int rm   = lane & 15;

  v8f zero = {};
  v8f acc0 = zero, acc1 = zero, acc2 = zero, acc3 = zero;

  for (int k0 = 0; k0 < K; k0 += 32) {
    __syncthreads();   // previous tile fully consumed before overwrite

    // ---- Stage A tile: 128 rows x 32 cols f32, async copy global -> LDS
    {
      int row = tid >> 1;
      int cb  = (tid & 1) * 16;
      const float* src = A + (size_t)(m0 + row) * K + k0 + cb;
      unsigned ldsoff = (unsigned)(size_t)(&lAf[row][cb]);
#if defined(__HIP_DEVICE_COMPILE__)
      #pragma unroll
      for (int u = 0; u < 16; u += 4) {
        unsigned long ga = (unsigned long)(src + u);
        unsigned lo = ldsoff + u * 4;
        asm volatile("global_load_async_to_lds_b128 %0, %1, off"
                     :: "v"(lo), "v"(ga) : "memory");
      }
#else
      for (int u = 0; u < 16; u++) lAf[row][cb + u] = src[u];
#endif
    }
    // ---- Stage W tile: 64 rows x 32 cols, f32 -> f16 once (rows >= N -> 0)
    {
      int row = tid >> 2;
      int cb  = (tid & 3) * 8;
      int n   = n0 + row;
      _Float16* dst = &lBh[row][cb];
      if (n < N) {
        const float* srcb = W + (size_t)n * K + k0 + cb;
        #pragma unroll
        for (int u = 0; u < 8; u += 4) {
          float4 v = *(const float4*)(srcb + u);
          h2v p0; p0[0] = (_Float16)v.x; p0[1] = (_Float16)v.y;
          h2v p1; p1[0] = (_Float16)v.z; p1[1] = (_Float16)v.w;
          *(h2v*)(dst + u)     = p0;
          *(h2v*)(dst + u + 2) = p1;
        }
        if (k0 + 32 < K) __builtin_prefetch(srcb + 32, 0, 1);
      } else {
        #pragma unroll
        for (int u = 0; u < 8; u++) dst[u] = (_Float16)0.f;
      }
    }
#if defined(__HIP_DEVICE_COMPILE__)
    asm volatile("s_wait_asynccnt 0x0" ::: "memory");
#endif
    __syncthreads();

    // ---- A fragment (16x32 f16 layout): convert from f32 LDS
    v16h afrag;
    #pragma unroll
    for (int e = 0; e < 16; e++) {
      int kof = ((e & 8) << 1) + (hlf << 3) + (e & 7);
      afrag[e] = (_Float16)lAf[wave * 16 + rm][kof];
    }
    #pragma unroll
    for (int nb = 0; nb < 4; nb++) {
      // B fragment (32x16 f16 layout): packed f16 pairs straight from LDS
      v16h bfrag;
      #pragma unroll
      for (int e2 = 0; e2 < 8; e2++) {
        int kof = ((e2 & 4) << 2) + (hlf << 3) + (e2 & 3) * 2;
        h2v hv = *(const h2v*)&lBh[nb * 16 + rm][kof];
        bfrag[e2 * 2]     = hv[0];
        bfrag[e2 * 2 + 1] = hv[1];
      }
      v8f* accp = nb == 0 ? &acc0 : nb == 1 ? &acc1 : nb == 2 ? &acc2 : &acc3;
      *accp = __builtin_amdgcn_wmma_f32_16x16x32_f16(
          false, afrag, false, bfrag, (short)0, *accp, false, false);
    }
  }

  // Epilogue: D layout — reg r -> m = r + 8*hlf ; n = rm
  #pragma unroll
  for (int nb = 0; nb < 4; nb++) {
    int n = n0 + nb * 16 + rm;
    if (n >= N) continue;
    float bv = bias ? bias[n] : 0.f;
    v8f acc = nb == 0 ? acc0 : nb == 1 ? acc1 : nb == 2 ? acc2 : acc3;
    #pragma unroll
    for (int r = 0; r < 8; r++) {
      int m = m0 + wave * 16 + r + hlf * 8;
      C[(size_t)m * N + n] = acc[r] + bv;
    }
  }
}

// ---------------------------------------------------------------------------
// Token projections: glob = global_token @ w_glob^T + b_glob          (8,64)
//                    ttv  = silu(time_token) @ w_ssmt^T + b_ssmt      (8,256)
//                    tms  = silu(time_token) @ tm_w^T + tm_b          (8,128)
// ---------------------------------------------------------------------------
__global__ void tokens_kernel(
    const float* __restrict__ gt, const float* __restrict__ tt_in,
    const float* __restrict__ w_glob, const float* __restrict__ b_glob,
    const float* __restrict__ w_ssmt, const float* __restrict__ b_ssmt,
    const float* __restrict__ tm_w,  const float* __restrict__ tm_b,
    float* __restrict__ glob, float* __restrict__ ttv, float* __restrict__ tms)
{
  int b = blockIdx.x;
  __shared__ float st[256];
  for (int i = threadIdx.x; i < 256; i += blockDim.x) st[i] = siluf_(tt_in[b * 256 + i]);
  __syncthreads();
  for (int j = threadIdx.x; j < 64; j += blockDim.x) {
    float a = b_glob[j];
    for (int k = 0; k < 256; k++) a += gt[b * 256 + k] * w_glob[j * 256 + k];
    glob[b * 64 + j] = a;
  }
  for (int j = threadIdx.x; j < 256; j += blockDim.x) {
    float a = b_ssmt[j];
    for (int k = 0; k < 256; k++) a += st[k] * w_ssmt[j * 256 + k];
    ttv[b * 256 + j] = a;
  }
  for (int j = threadIdx.x; j < 128; j += blockDim.x) {
    float a = tm_b[j];
    for (int k = 0; k < 256; k++) a += st[k] * tm_w[j * 256 + k];
    tms[b * 128 + j] = a;
  }
}

// ---------------------------------------------------------------------------
// LayerNorm over (C,H,W) per batch: stats then apply with cond modulation.
// ---------------------------------------------------------------------------
__global__ void ln_stats_kernel(const float* __restrict__ x, float* __restrict__ st)
{
  int b = blockIdx.x;
  const float* xb = x + (size_t)b * 65536;
  __shared__ float r1[256], r2[256];
  float s = 0, q = 0;
  for (int i = threadIdx.x; i < 65536; i += 256) { float v = xb[i]; s += v; q += v * v; }
  r1[threadIdx.x] = s; r2[threadIdx.x] = q;
  __syncthreads();
  for (int t = 128; t > 0; t >>= 1) {
    if (threadIdx.x < t) { r1[threadIdx.x] += r1[threadIdx.x + t]; r2[threadIdx.x] += r2[threadIdx.x + t]; }
    __syncthreads();
  }
  if (threadIdx.x == 0) {
    float m = r1[0] / 65536.f;
    float v = r2[0] / 65536.f - m * m;
    st[b * 2] = m; st[b * 2 + 1] = rsqrtf(v + 1e-6f);
  }
}

// x NCHW -> out NHWC, modulated by cond_base channels [sc_off]/[sh_off]
__global__ void ln_apply_kernel(const float* __restrict__ x, const float* __restrict__ cond,
                                const float* __restrict__ st, float* __restrict__ out,
                                int sc_off, int sh_off)
{
  int tid = blockIdx.x * 256 + threadIdx.x;   // over 8*64*1024 (NCHW order)
  int p = tid & 1023, c = (tid >> 10) & 63, b = tid >> 16;
  float m = st[b * 2], r = st[b * 2 + 1];
  float v = (x[tid] - m) * r;
  float sc = cond[((size_t)b * 512 + sc_off + c) * 1024 + p];
  float sh = cond[((size_t)b * 512 + sh_off + c) * 1024 + p];
  out[((size_t)(b * 1024 + p)) * 64 + c] = v * (1.f + sc) + sh;
}

// ---------------------------------------------------------------------------
// Gate branch: x_gate = silu(xp[...,128:] + gamma_gate * cond_bias_tiled)
// ---------------------------------------------------------------------------
__global__ void gate_kernel(const float* __restrict__ xp, const float* __restrict__ cond,
                            const float* __restrict__ glob, const float* __restrict__ ggate,
                            float* __restrict__ gate)
{
  int tid = blockIdx.x * 256 + threadIdx.x;   // 8192*128
  int c = tid & 127, pix = tid >> 7, b = pix >> 10, p = pix & 1023;
  int cc = c & 63;
  float cb = cond[((size_t)b * 512 + 384 + cc) * 1024 + p] + glob[b * 64 + cc];
  float g = xp[(size_t)pix * 256 + 128 + c] + ggate[0] * cb;
  gate[tid] = siluf_(g);
}

// ---------------------------------------------------------------------------
// Depthwise 3x3 on xp[...,:128] + silu + cond/time modulation -> xss (NHWC 128)
// ---------------------------------------------------------------------------
__global__ void dwconv_mod_kernel(const float* __restrict__ xp, const float* __restrict__ w_dw,
                                  const float* __restrict__ b_dw, const float* __restrict__ cond,
                                  const float* __restrict__ glob, const float* __restrict__ gmain,
                                  const float* __restrict__ ttv, float* __restrict__ xss)
{
  int tid = blockIdx.x * 256 + threadIdx.x;   // 8192*128
  int c = tid & 127, pix = tid >> 7, b = pix >> 10, p = pix & 1023;
  int y = p >> 5, x = p & 31;
  float acc = b_dw[c];
  #pragma unroll
  for (int ky = 0; ky < 3; ky++) {
    int yy = y + ky - 1; if (yy < 0 || yy > 31) continue;
    #pragma unroll
    for (int kx = 0; kx < 3; kx++) {
      int xx = x + kx - 1; if (xx < 0 || xx > 31) continue;
      acc += xp[((size_t)(b * 1024 + yy * 32 + xx)) * 256 + c] * w_dw[c * 9 + ky * 3 + kx];
    }
  }
  float v = siluf_(acc);
  int cc = c & 63;
  float cb = cond[((size_t)b * 512 + 384 + cc) * 1024 + p] + glob[b * 64 + cc];
  v = (v + gmain[0] * cb) * (1.f + ttv[b * 256 + c]) + ttv[b * 256 + 128 + c];
  xss[tid] = v;
}

// ---------------------------------------------------------------------------
// Pack 4 scan directions: seq[dir,b,l,c] = xss[b, map(dir,l), c]
// ---------------------------------------------------------------------------
__global__ void pack_kernel(const float* __restrict__ xss, float* __restrict__ seq)
{
  int tid = blockIdx.x * 256 + threadIdx.x;   // 4*8*1024*128
  int c = tid & 127, l = (tid >> 7) & 1023, b = (tid >> 17) & 7, dir = tid >> 20;
  int tau = ((l & 31) << 5) | (l >> 5);
  int hw = (dir == 0) ? l : (dir == 1) ? (1023 - l) : (dir == 2) ? tau : (1023 - tau);
  seq[tid] = xss[((size_t)(b * 1024 + hw)) * 128 + c];
}

// ---------------------------------------------------------------------------
// Causal depthwise conv1d (k=4) over sequence + silu  -> xmc
// ---------------------------------------------------------------------------
__global__ void conv1d_kernel(const float* __restrict__ xz, const float* __restrict__ w,
                              const float* __restrict__ bias, float* __restrict__ xmc)
{
  int tid = blockIdx.x * 256 + threadIdx.x;   // 32768*256
  int d = tid & 255, l = (tid >> 8) & 1023, s = tid >> 18;
  float acc = bias[d];
  #pragma unroll
  for (int t = 0; t < 4; t++) {
    int ll = l - 3 + t;
    if (ll >= 0) acc += w[d * 4 + t] * xz[((size_t)(s * 1024 + ll)) * 512 + d];
  }
  xmc[tid] = siluf_(acc);
}

// ---------------------------------------------------------------------------
// dt = softplus(x_dbl[:, :8] @ m_dt_w^T + m_dt_b)
// ---------------------------------------------------------------------------
__global__ void dt_kernel(const float* __restrict__ xdbl, const float* __restrict__ w,
                          const float* __restrict__ bias, float* __restrict__ dtb)
{
  int tid = blockIdx.x * 256 + threadIdx.x;   // 32768*256
  int j = tid & 255, row = tid >> 8;
  float a = bias[j];
  #pragma unroll
  for (int k = 0; k < 8; k++) a += xdbl[(size_t)row * 40 + k] * w[j * 8 + k];
  dtb[tid] = (a > 20.f) ? a : log1pf(__expf(a));
}

// ---------------------------------------------------------------------------
// Selective scan: block = one (dir,batch), thread = channel d; 1024 steps.
// Output folds +x*D and *silu(z).
// ---------------------------------------------------------------------------
__global__ __launch_bounds__(256) void scan_kernel(
    const float* __restrict__ xdbl, const float* __restrict__ dtb,
    const float* __restrict__ xmc,  const float* __restrict__ xz,
    const float* __restrict__ Alog, const float* __restrict__ Dp,
    float* __restrict__ yfin)
{
  int s = blockIdx.x;       // 0..31
  int d = threadIdx.x;      // 0..255
  float A[16], h[16];
  #pragma unroll
  for (int n = 0; n < 16; n++) { A[n] = -__expf(Alog[d * 16 + n]); h[n] = 0.f; }
  float Dd = Dp[d];
  __shared__ float Bs[16], Cs[16];
  for (int l = 0; l < 1024; l++) {
    size_t row = (size_t)s * 1024 + l;
    __syncthreads();
    if (threadIdx.x < 16)      Bs[threadIdx.x]      = xdbl[row * 40 + 8  + threadIdx.x];
    else if (threadIdx.x < 32) Cs[threadIdx.x - 16] = xdbl[row * 40 + 24 + (threadIdx.x - 16)];
    __syncthreads();
    float dt = dtb[row * 256 + d];
    float xv = xmc[row * 256 + d];
    float y = 0.f;
    #pragma unroll
    for (int n = 0; n < 16; n++) {
      h[n] = __expf(dt * A[n]) * h[n] + (dt * Bs[n]) * xv;
      y += h[n] * Cs[n];
    }
    float zv = xz[row * 512 + 256 + d];
    yfin[row * 256 + d] = (y + xv * Dd) * siluf_(zv);
  }
}

// ---------------------------------------------------------------------------
// Merge 4 directions back to image + gate multiply -> x_fused (NHWC 128)
// ---------------------------------------------------------------------------
__global__ void merge_kernel(const float* __restrict__ oseq, const float* __restrict__ gate,
                             float* __restrict__ xfused)
{
  int tid = blockIdx.x * 256 + threadIdx.x;   // 8192*128
  int c = tid & 127, p = (tid >> 7) & 1023, b = tid >> 17;
  int tau = ((p & 31) << 5) | (p >> 5);
  float sum =
      oseq[(((size_t)(0 * 8 + b) * 1024) + p) * 128 + c] +
      oseq[(((size_t)(1 * 8 + b) * 1024) + (1023 - p)) * 128 + c] +
      oseq[(((size_t)(2 * 8 + b) * 1024) + tau) * 128 + c] +
      oseq[(((size_t)(3 * 8 + b) * 1024) + (1023 - tau)) * 128 + c];
  xfused[tid] = sum * gate[tid];
}

// ---------------------------------------------------------------------------
// Physics prior stage 1: Gaussian blurs (reflect pad), normalized residuals,
// energy map.  dsdl: (8,1024,2) NHWC ; e: (8,1024)
// ---------------------------------------------------------------------------
__device__ __forceinline__ int refl_(int i) { return i < 0 ? -i : (i > 31 ? 62 - i : i); }

__global__ __launch_bounds__(256) void physics_pan_kernel(
    const float* __restrict__ pan, float* __restrict__ dsdl, float* __restrict__ e)
{
  int b = blockIdx.x;
  int tid = threadIdx.x;
  const float* pb = pan + (size_t)b * 1024;
  float ks[9], kl[25];
  {
    float s = 0;
    for (int i = 0; i < 9; i++) { int dy = i / 3 - 1, dx = i % 3 - 1;
      float g = __expf(-0.5f * (float)(dx * dx + dy * dy)); ks[i] = g; s += g; }
    for (int i = 0; i < 9; i++) ks[i] /= s;
  }
  {
    float s = 0;
    for (int i = 0; i < 25; i++) { int dy = i / 5 - 2, dx = i % 5 - 2;
      float g = __expf(-(float)(dx * dx + dy * dy) / 8.0f); kl[i] = g; s += g; }
    for (int i = 0; i < 25; i++) kl[i] /= s;
  }
  __shared__ float r1[256], r2[256], r3[256], r4[256];
  float s1 = 0, q1 = 0, s2 = 0, q2 = 0;
  for (int p = tid; p < 1024; p += 256) {
    int y = p >> 5, x = p & 31;
    float bs = 0;
    for (int i = 0; i < 9; i++)  bs += ks[i] * pb[refl_(y + i / 3 - 1) * 32 + refl_(x + i % 3 - 1)];
    float bl = 0;
    for (int i = 0; i < 25; i++) bl += kl[i] * pb[refl_(y + i / 5 - 2) * 32 + refl_(x + i % 5 - 2)];
    float v = pb[p];
    float a = v - bs, c = v - bl;
    dsdl[((size_t)(b * 1024 + p)) * 2 + 0] = a;
    dsdl[((size_t)(b * 1024 + p)) * 2 + 1] = c;
    s1 += a; q1 += a * a; s2 += c; q2 += c * c;
  }
  r1[tid] = s1; r2[tid] = q1; r3[tid] = s2; r4[tid] = q2;
  __syncthreads();
  for (int t = 128; t > 0; t >>= 1) {
    if (tid < t) { r1[tid] += r1[tid + t]; r2[tid] += r2[tid + t];
                   r3[tid] += r3[tid + t]; r4[tid] += r4[tid + t]; }
    __syncthreads();
  }
  float sig_s = sqrtf(fmaxf((r2[0] - r1[0] * r1[0] / 1024.f) / 1023.f, 0.f)) + 1e-6f;
  float sig_l = sqrtf(fmaxf((r4[0] - r3[0] * r3[0] / 1024.f) / 1023.f, 0.f)) + 1e-6f;
  __syncthreads();
  float mx = 0;
  for (int p = tid; p < 1024; p += 256) {
    size_t ix = ((size_t)(b * 1024 + p)) * 2;
    float a = dsdl[ix + 0] / sig_s;
    float c = dsdl[ix + 1] / sig_l;
    dsdl[ix + 0] = a; dsdl[ix + 1] = c;
    float er = fabsf(a) + fabsf(c);
    e[b * 1024 + p] = er;
    mx = fmaxf(mx, er);
  }
  r1[tid] = mx;
  __syncthreads();
  for (int t = 128; t > 0; t >>= 1) {
    if (tid < t) r1[tid] = fmaxf(r1[tid], r1[tid + t]);
    __syncthreads();
  }
  float M = r1[0] + 1e-6f;
  for (int p = tid; p < 1024; p += 256) e[b * 1024 + p] /= M;
}

// ---------------------------------------------------------------------------
// Direct 3x3 conv (zero pad).  in NHWC (or NCHW if nchw_in), out NHWC.
// grid = (B*1024/256, Co); weights for one co staged in LDS.
// ---------------------------------------------------------------------------
__global__ void conv3x3_kernel(const float* __restrict__ in, const float* __restrict__ w,
                               const float* __restrict__ bias, const float* __restrict__ resid,
                               float* __restrict__ out, int Cin, int Co, int nchw_in)
{
  __shared__ float lw[576];
  int co = blockIdx.y;
  for (int i = threadIdx.x; i < Cin * 9; i += 256) lw[i] = w[(size_t)co * Cin * 9 + i];
  __syncthreads();
  int pix = blockIdx.x * 256 + threadIdx.x;   // 0..8191
  int b = pix >> 10, p = pix & 1023;
  int y = p >> 5, x = p & 31;
  float acc = bias ? bias[co] : 0.f;
  for (int ky = 0; ky < 3; ky++) {
    int yy = y + ky - 1; if (yy < 0 || yy > 31) continue;
    for (int kx = 0; kx < 3; kx++) {
      int xx = x + kx - 1; if (xx < 0 || xx > 31) continue;
      int q = yy * 32 + xx;
      if (nchw_in) {
        for (int ci = 0; ci < Cin; ci++)
          acc += in[((size_t)(b * Cin + ci)) * 1024 + q] * lw[ci * 9 + ky * 3 + kx];
      } else {
        const float* ib = in + ((size_t)(b * 1024 + q)) * Cin;
        for (int ci = 0; ci < Cin; ci++)
          acc += ib[ci] * lw[ci * 9 + ky * 3 + kx];
      }
    }
  }
  if (resid) acc += resid[((size_t)(b * 1024 + p)) * Co + co];
  out[((size_t)(b * 1024 + p)) * Co + co] = acc;
}

// ---------------------------------------------------------------------------
// GroupNorm (NHWC, 1024 pixels) with optional affine / per-(b,c) modulation /
// silu.  grid = B*groups blocks.
// ---------------------------------------------------------------------------
__global__ void groupnorm_kernel(const float* __restrict__ in, float* __restrict__ out,
                                 const float* __restrict__ gamma, const float* __restrict__ beta,
                                 const float* __restrict__ mod, int C, int groups, int do_silu)
{
  int bg = blockIdx.x;
  int b = bg / groups, g = bg % groups;
  int cpg = C / groups;
  int nel = cpg * 1024;
  __shared__ float r1[256], r2[256];
  float s = 0, q = 0;
  for (int i = threadIdx.x; i < nel; i += 256) {
    int c = g * cpg + i / 1024, p = i & 1023;
    float v = in[((size_t)(b * 1024 + p)) * C + c];
    s += v; q += v * v;
  }
  r1[threadIdx.x] = s; r2[threadIdx.x] = q;
  __syncthreads();
  for (int t = 128; t > 0; t >>= 1) {
    if (threadIdx.x < t) { r1[threadIdx.x] += r1[threadIdx.x + t]; r2[threadIdx.x] += r2[threadIdx.x + t]; }
    __syncthreads();
  }
  float m = r1[0] / nel;
  float var = r2[0] / nel - m * m;
  float rstd = rsqrtf(var + 1e-5f);
  for (int i = threadIdx.x; i < nel; i += 256) {
    int c = g * cpg + i / 1024, p = i & 1023;
    size_t ix = ((size_t)(b * 1024 + p)) * C + c;
    float v = (in[ix] - m) * rstd;
    if (gamma) v = v * gamma[c] + beta[c];
    if (mod)   v = v * (1.f + mod[(size_t)b * 2 * C + c]) + mod[(size_t)b * 2 * C + C + c];
    if (do_silu) v = siluf_(v);
    out[ix] = v;
  }
}

// ---------------------------------------------------------------------------
// Build 192-channel concat for sf1: [phys | tanh(alpha)*learned | e*en_w+en_b]
// ---------------------------------------------------------------------------
__global__ void build_cat_kernel(const float* __restrict__ phys, const float* __restrict__ learned,
                                 const float* __restrict__ e, const float* __restrict__ alpha,
                                 const float* __restrict__ en_w, const float* __restrict__ en_b,
                                 float* __restrict__ cat)
{
  int tid = blockIdx.x * 256 + threadIdx.x;   // 8192*192
  int k = tid % 192, pix = tid / 192;
  float v;
  if (k < 64)       v = phys[(size_t)pix * 64 + k];
  else if (k < 128) v = tanhf(alpha[0]) * learned[(size_t)pix * 64 + (k - 64)];
  else              v = e[pix] * en_w[k - 128] + en_b[k - 128];
  cat[tid] = v;
}

// ---------------------------------------------------------------------------
// Apply physics scale/shift gating to x_fused -> xo_in (NHWC 128)
// ---------------------------------------------------------------------------
__global__ void apply_ph_kernel(const float* __restrict__ ph, const float* __restrict__ xfused,
                                float* __restrict__ xoin)
{
  int tid = blockIdx.x * 256 + threadIdx.x;   // 8192*128
  int d = tid & 127, pix = tid >> 7;
  const float* pr = ph + (size_t)pix * 384;
  float g = 0.3f + 0.7f * sigmoidf_(pr[256 + d]);
  xoin[tid] = xfused[tid] * (1.f + pr[d] * g) + pr[128 + d] * g;
}

// xx = shortcut + xo * sm_gate   (NCHW)
__global__ void resid1_kernel(const float* __restrict__ x, const float* __restrict__ xo,
                              const float* __restrict__ cond, float* __restrict__ xx)
{
  int tid = blockIdx.x * 256 + threadIdx.x;   // 8*64*1024 (NCHW)
  int p = tid & 1023, c = (tid >> 10) & 63, b = tid >> 16;
  xx[tid] = x[tid] + xo[((size_t)(b * 1024 + p)) * 64 + c] *
                         cond[((size_t)b * 512 + 128 + c) * 1024 + p];
}

__global__ void gelu_kernel(float* __restrict__ h)
{
  int tid = blockIdx.x * 256 + threadIdx.x;
  float v = h[tid];
  h[tid] = 0.5f * v * (1.f + erff(v * 0.70710678118654752f));
}

// out = xx + mlp * sp_gate   (NCHW)
__global__ void final_kernel(const float* __restrict__ xx, const float* __restrict__ h2,
                             const float* __restrict__ cond, float* __restrict__ out)
{
  int tid = blockIdx.x * 256 + threadIdx.x;   // 8*64*1024 (NCHW)
  int p = tid & 1023, c = (tid >> 10) & 63, b = tid >> 16;
  out[tid] = xx[tid] + h2[((size_t)(b * 1024 + p)) * 64 + c] *
                           cond[((size_t)b * 512 + 320 + c) * 1024 + p];
}

// ---------------------------------------------------------------------------
// Host launcher
// ---------------------------------------------------------------------------
extern "C" void kernel_launch(void* const* d_in, const int* in_sizes, int n_in,
                              void* d_out, int out_size, void* d_ws, size_t ws_size,
                              hipStream_t stream)
{
  const float* x          = (const float*)d_in[0];
  const float* cond_base  = (const float*)d_in[1];
  const float* cond_hf    = (const float*)d_in[2];
  const float* pan_raw    = (const float*)d_in[3];
  const float* global_tok = (const float*)d_in[4];
  const float* time_tok   = (const float*)d_in[5];
  const float* w_in       = (const float*)d_in[6];
  const float* b_in       = (const float*)d_in[7];
  const float* w_dw       = (const float*)d_in[8];
  const float* b_dw       = (const float*)d_in[9];
  const float* w_glob     = (const float*)d_in[10];
  const float* b_glob     = (const float*)d_in[11];
  const float* w_ssmt     = (const float*)d_in[12];
  const float* b_ssmt     = (const float*)d_in[13];
  const float* gamma_gate = (const float*)d_in[14];
  const float* gamma_main = (const float*)d_in[15];
  const float* w_out      = (const float*)d_in[16];
  const float* b_out      = (const float*)d_in[17];
  const float* w_fc1      = (const float*)d_in[18];
  const float* b_fc1      = (const float*)d_in[19];
  const float* w_fc2      = (const float*)d_in[20];
  const float* b_fc2      = (const float*)d_in[21];
  const float* m_in_w     = (const float*)d_in[22];
  const float* m_conv_w   = (const float*)d_in[23];
  const float* m_conv_b   = (const float*)d_in[24];
  const float* m_xproj_w  = (const float*)d_in[25];
  const float* m_dt_w     = (const float*)d_in[26];
  const float* m_dt_b     = (const float*)d_in[27];
  const float* m_Alog     = (const float*)d_in[28];
  const float* m_D        = (const float*)d_in[29];
  const float* m_out_w    = (const float*)d_in[30];
  const float* pp1_w      = (const float*)d_in[31];
  const float* pp1_b      = (const float*)d_in[32];
  const float* pp_g1      = (const float*)d_in[33];
  const float* pp_b1      = (const float*)d_in[34];
  const float* pp2_w      = (const float*)d_in[35];
  const float* pp2_b      = (const float*)d_in[36];
  const float* al_w       = (const float*)d_in[37];
  const float* al_b       = (const float*)d_in[38];
  const float* alpha_aux  = (const float*)d_in[39];
  const float* en_w       = (const float*)d_in[40];
  const float* en_b       = (const float*)d_in[41];
  const float* sf1_w      = (const float*)d_in[42];
  const float* sf_g1      = (const float*)d_in[43];
  const float* sf_b1      = (const float*)d_in[44];
  const float* sf2_w      = (const float*)d_in[45];
  const float* sf_g2      = (const float*)d_in[46];
  const float* sf_b2      = (const float*)d_in[47];
  const float* sf3_w      = (const float*)d_in[48];
  const float* sf3_b      = (const float*)d_in[49];
  const float* tm_w       = (const float*)d_in[50];
  const float* tm_b       = (const float*)d_in[51];
  const float* ph_w       = (const float*)d_in[52];
  const float* ph_b       = (const float*)d_in[53];
  float* out = (float*)d_out;

  // workspace allocator (floats, 256-float aligned)
  float* Wb = (float*)d_ws;
  size_t off = 0;
  auto alloc = [&](size_t n) -> float* {
    float* p = Wb + off;
    off += (n + 255) & ~(size_t)255;
    return p;
  };

  float* glob   = alloc(512);
  float* ttv    = alloc(2048);
  float* tms    = alloc(1024);
  float* st1    = alloc(16);
  float* st2    = alloc(16);
  float* xnorm  = alloc(8192ull * 64);
  float* xp     = alloc(8192ull * 256);
  float* gate   = alloc(8192ull * 128);
  float* xss    = alloc(8192ull * 128);
  float* seq    = alloc(32768ull * 128);
  float* xz     = alloc(32768ull * 512);
  float* xmc    = alloc(32768ull * 256);
  float* xdbl   = alloc(32768ull * 40);
  float* dtb    = alloc(32768ull * 256);
  float* yfin   = alloc(32768ull * 256);
  float* oseq   = alloc(32768ull * 128);
  float* xfused = alloc(8192ull * 128);
  float* dsdl   = alloc(8192ull * 2);
  float* ebuf   = alloc(8192);
  float* fpre   = alloc(8192ull * 64);
  float* phys   = alloc(8192ull * 64);
  float* learned= alloc(8192ull * 64);
  float* catb   = alloc(8192ull * 192);
  float* s1     = alloc(8192ull * 64);
  float* s2     = alloc(8192ull * 64);
  float* dseed  = alloc(8192ull * 64);
  float* abuf   = alloc(8192ull * 64);
  float* phb    = alloc(8192ull * 384);
  float* xoin   = alloc(8192ull * 128);
  float* xo     = alloc(8192ull * 64);
  float* xxb    = alloc(8192ull * 64);
  float* xm2    = alloc(8192ull * 64);
  float* h1     = alloc(8192ull * 256);
  float* h2     = alloc(8192ull * 64);
  (void)ws_size; (void)n_in; (void)in_sizes; (void)out_size;

  // ---- tokens & LN1
  tokens_kernel<<<8, 256, 0, stream>>>(global_tok, time_tok, w_glob, b_glob,
                                       w_ssmt, b_ssmt, tm_w, tm_b, glob, ttv, tms);
  ln_stats_kernel<<<8, 256, 0, stream>>>(x, st1);
  ln_apply_kernel<<<2048, 256, 0, stream>>>(x, cond_base, st1, xnorm, /*sc*/64, /*sh*/0);

  // ---- in-proj GEMM: xp = xnorm @ w_in^T + b_in   (8192x64 -> 256)
  gemm_wmma_kernel<<<dim3(64, 4), 256, 0, stream>>>(xnorm, w_in, b_in, xp, 8192, 64, 256);

  // ---- gate branch & SS2D input branch
  gate_kernel<<<4096, 256, 0, stream>>>(xp, cond_base, glob, gamma_gate, gate);
  dwconv_mod_kernel<<<4096, 256, 0, stream>>>(xp, w_dw, b_dw, cond_base, glob,
                                              gamma_main, ttv, xss);
  pack_kernel<<<16384, 256, 0, stream>>>(xss, seq);

  // ---- mamba in-proj: xz = seq @ m_in_w^T   (32768x128 -> 512)
  gemm_wmma_kernel<<<dim3(256, 8), 256, 0, stream>>>(seq, m_in_w, nullptr, xz, 32768, 128, 512);

  // ---- causal conv1d + silu
  conv1d_kernel<<<32768, 256, 0, stream>>>(xz, m_conv_w, m_conv_b, xmc);

  // ---- x_dbl = xmc @ m_xproj_w^T   (32768x256 -> 40)
  gemm_wmma_kernel<<<dim3(256, 1), 256, 0, stream>>>(xmc, m_xproj_w, nullptr, xdbl, 32768, 256, 40);

  // ---- dt = softplus(...)
  dt_kernel<<<32768, 256, 0, stream>>>(xdbl, m_dt_w, m_dt_b, dtb);

  // ---- selective scan (folds +x*D and *silu(z))
  scan_kernel<<<32, 256, 0, stream>>>(xdbl, dtb, xmc, xz, m_Alog, m_D, yfin);

  // ---- out-proj: oseq = yfin @ m_out_w^T   (32768x256 -> 128)
  gemm_wmma_kernel<<<dim3(256, 2), 256, 0, stream>>>(yfin, m_out_w, nullptr, oseq, 32768, 256, 128);

  // ---- merge 4 directions + gate
  merge_kernel<<<4096, 256, 0, stream>>>(oseq, gate, xfused);

  // ---- physics prior chain
  physics_pan_kernel<<<8, 256, 0, stream>>>(pan_raw, dsdl, ebuf);
  conv3x3_kernel<<<dim3(32, 64), 256, 0, stream>>>(dsdl, pp1_w, pp1_b, nullptr, fpre, 2, 64, 0);
  groupnorm_kernel<<<256, 256, 0, stream>>>(fpre, fpre, pp_g1, pp_b1, nullptr, 64, 32, 1);
  conv3x3_kernel<<<dim3(32, 64), 256, 0, stream>>>(fpre, pp2_w, pp2_b, nullptr, phys, 64, 64, 0);
  conv3x3_kernel<<<dim3(32, 64), 256, 0, stream>>>(cond_hf, al_w, al_b, nullptr, learned, 64, 64, 1);
  build_cat_kernel<<<6144, 256, 0, stream>>>(phys, learned, ebuf, alpha_aux, en_w, en_b, catb);
  gemm_wmma_kernel<<<dim3(64, 1), 256, 0, stream>>>(catb, sf1_w, nullptr, s1, 8192, 192, 64);
  groupnorm_kernel<<<256, 256, 0, stream>>>(s1, s1, sf_g1, sf_b1, nullptr, 64, 32, 1);
  conv3x3_kernel<<<dim3(32, 64), 256, 0, stream>>>(s1, sf2_w, nullptr, nullptr, s2, 64, 64, 0);
  groupnorm_kernel<<<256, 256, 0, stream>>>(s2, s2, sf_g2, sf_b2, nullptr, 64, 32, 1);
  conv3x3_kernel<<<dim3(32, 64), 256, 0, stream>>>(s2, sf3_w, sf3_b, phys, dseed, 64, 64, 0);
  groupnorm_kernel<<<256, 256, 0, stream>>>(dseed, abuf, nullptr, nullptr, tms, 64, 32, 1);
  gemm_wmma_kernel<<<dim3(64, 6), 256, 0, stream>>>(abuf, ph_w, ph_b, phb, 8192, 64, 384);
  apply_ph_kernel<<<4096, 256, 0, stream>>>(phb, xfused, xoin);

  // ---- out proj + residual 1
  gemm_wmma_kernel<<<dim3(64, 1), 256, 0, stream>>>(xoin, w_out, b_out, xo, 8192, 128, 64);
  resid1_kernel<<<2048, 256, 0, stream>>>(x, xo, cond_base, xxb);

  // ---- LN2 + MLP + residual 2
  ln_stats_kernel<<<8, 256, 0, stream>>>(xxb, st2);
  ln_apply_kernel<<<2048, 256, 0, stream>>>(xxb, cond_base, st2, xm2, /*sc*/256, /*sh*/192);
  gemm_wmma_kernel<<<dim3(64, 4), 256, 0, stream>>>(xm2, w_fc1, b_fc1, h1, 8192, 64, 256);
  gelu_kernel<<<8192, 256, 0, stream>>>(h1);
  gemm_wmma_kernel<<<dim3(64, 1), 256, 0, stream>>>(h1, w_fc2, b_fc2, h2, 8192, 256, 64);
  final_kernel<<<2048, 256, 0, stream>>>(xxb, h2, cond_base, out);
}
